// SelfAttention6_24970939859702
// MI455X (gfx1250) — compile-verified, matrix operand order checked by reading, not run on hardware
//
#include <hip/hip_runtime.h>
#include <hip/hip_bf16.h>
#include <math.h>

// ---------------- problem dims ----------------
#define BB 64
#define SS 1024
#define HH 512
#define AA 256
#define FF 2048
#define ROWS ((size_t)BB * SS)   // 65536 tokens

typedef __attribute__((ext_vector_type(16))) __bf16 v16bf;
typedef __attribute__((ext_vector_type(8)))  __bf16 v8bf;
typedef __attribute__((ext_vector_type(8)))  float  v8f;
typedef __bf16 __attribute__((may_alias)) bf16_alias;

// ---------------- WMMA helpers (CDNA5 16x16x32 bf16, ISA 7.12.2 layouts) ----
__device__ __forceinline__ v8f zero8() {
  v8f z;
#pragma unroll
  for (int i = 0; i < 8; ++i) z[i] = 0.0f;
  return z;
}

__device__ __forceinline__ v8f wmma_bf16(v16bf a, v16bf b, v8f c) {
  // (neg_a, A, neg_b, B, c_mod, C, reuse_a, reuse_b)
  return __builtin_amdgcn_wmma_f32_16x16x32_bf16(false, a, false, b,
                                                 (short)0, c, false, false);
}

// 16x32 fragment from a row-major [R x K] bf16 matrix, rows r0..r0+15,
// k-cols k0..k0+31.  Per ISA layout the per-lane data is exactly two
// contiguous 8-element runs -> two 16-byte loads, zero repacking.
// Used for A operands and for B operands stored K-contiguous (Bt[n][k]).
__device__ __forceinline__ v16bf load_frag16(const __bf16* M, int ld,
                                             int r0, int k0, int lane) {
  const __bf16* p = M + (size_t)(r0 + (lane & 15)) * ld + k0 + ((lane & 16) ? 8 : 0);
  v8bf lo = *(const v8bf*)p;         // k = kb .. kb+7
  v8bf hi = *(const v8bf*)(p + 16);  // k = 16+kb .. 16+kb+7
  return __builtin_shufflevector(lo, hi, 0, 1, 2, 3, 4, 5, 6, 7,
                                 8, 9, 10, 11, 12, 13, 14, 15);
}

// C/D store: VGPR r -> row m0 + r (+8 for lanes 16..31), col n0 + lane%16
__device__ __forceinline__ void store_c_bf16(__bf16* C, int ldc, int m0, int n0,
                                             int lane, v8f c) {
  int n  = n0 + (lane & 15);
  int mb = m0 + ((lane & 16) ? 8 : 0);
#pragma unroll
  for (int r = 0; r < 8; ++r) C[(size_t)(mb + r) * ldc + n] = (__bf16)c[r];
}

// 16x64 output tile over NT K-steps, explicitly double-buffered in registers:
// set-1 fragment loads are issued before the wmmas consuming set-0, so each
// wmma group only waits for its own fragments (partial loadcnt waits) while
// the next group's 10 b128 loads are in flight.  Fully unrolled: all
// addresses collapse to base + immediate offsets.
template <int NT>
__device__ __forceinline__ void gemm_tile(const __bf16* A, int lda, int ar0,
                                          const __bf16* B, int ldb, int br0,
                                          int lane, v8f* acc) {
  v16bf a0, b00, b01, b02, b03;
  v16bf a1, b10, b11, b12, b13;
  a0  = load_frag16(A, lda, ar0,      0, lane);
  b00 = load_frag16(B, ldb, br0 +  0, 0, lane);
  b01 = load_frag16(B, ldb, br0 + 16, 0, lane);
  b02 = load_frag16(B, ldb, br0 + 32, 0, lane);
  b03 = load_frag16(B, ldb, br0 + 48, 0, lane);
#pragma unroll
  for (int kt = 0; kt < NT; kt += 2) {
    if (kt + 1 < NT) {
      int k = (kt + 1) * 32;
      a1  = load_frag16(A, lda, ar0,      k, lane);
      b10 = load_frag16(B, ldb, br0 +  0, k, lane);
      b11 = load_frag16(B, ldb, br0 + 16, k, lane);
      b12 = load_frag16(B, ldb, br0 + 32, k, lane);
      b13 = load_frag16(B, ldb, br0 + 48, k, lane);
    }
    acc[0] = wmma_bf16(a0, b00, acc[0]);
    acc[1] = wmma_bf16(a0, b01, acc[1]);
    acc[2] = wmma_bf16(a0, b02, acc[2]);
    acc[3] = wmma_bf16(a0, b03, acc[3]);
    if (kt + 2 < NT) {
      int k = (kt + 2) * 32;
      a0  = load_frag16(A, lda, ar0,      k, lane);
      b00 = load_frag16(B, ldb, br0 +  0, k, lane);
      b01 = load_frag16(B, ldb, br0 + 16, k, lane);
      b02 = load_frag16(B, ldb, br0 + 32, k, lane);
      b03 = load_frag16(B, ldb, br0 + 48, k, lane);
    }
    if (kt + 1 < NT) {
      acc[0] = wmma_bf16(a1, b10, acc[0]);
      acc[1] = wmma_bf16(a1, b11, acc[1]);
      acc[2] = wmma_bf16(a1, b12, acc[2]);
      acc[3] = wmma_bf16(a1, b13, acc[3]);
    }
  }
}

// ---------------- kernel 0: f32 -> bf16 transposed conversion ----------------
// src row-major [K x N] f32  ->  dst row-major [N x K] bf16
__global__ void k_f32_to_bf16_T(const float* __restrict__ src,
                                __bf16* __restrict__ dst, int K, int N) {
  int i = blockIdx.x * blockDim.x + threadIdx.x;
  if (i < K * N) {
    int k = i / N, n = i % N;
    dst[(size_t)n * K + k] = (__bf16)src[i];
  }
}

// ---------------- kernel 1: embedding + projection + PE ----------------
__global__ void k_embed_proj(const int* __restrict__ cate_x,
                             const float* __restrict__ cont_x,
                             const float* __restrict__ E_cate,
                             const float* __restrict__ W_proj,
                             float* __restrict__ x_f32,
                             __bf16* __restrict__ x_b16) {
  __shared__ __align__(16) float in[32][264];  // 32 tokens x 260 features
  int row0 = blockIdx.x * 32;
  int tid  = threadIdx.x;

  for (int idx = tid; idx < 32 * 260; idx += 256) {
    int tok = idx / 260, f = idx % 260;
    int gr = row0 + tok;
    float v;
    if (f < 256) {
      int cat = f >> 6, e = f & 63;
      int vi = cate_x[(size_t)gr * 4 + cat];
      v = E_cate[((size_t)cat * 1000 + vi) * 64 + e];
    } else {
      v = cont_x[(size_t)gr * 4 + (f - 256)];
    }
    in[tok][f] = v;
  }
  __syncthreads();

  int tok   = tid >> 3;
  int hbase = (tid & 7) * 64;
  int gr    = row0 + tok;
  int s     = gr & (SS - 1);

  float acc[64];
#pragma unroll
  for (int j = 0; j < 64; ++j) acc[j] = 0.0f;

  for (int k = 0; k < 260; ++k) {
    float a = in[tok][k];
    const float* wr = W_proj + (size_t)k * HH + hbase;
#pragma unroll
    for (int j = 0; j < 64; ++j) acc[j] += a * wr[j];
  }

  float* xo = x_f32 + (size_t)gr * HH + hbase;
  __bf16* xb = x_b16 + (size_t)gr * HH + hbase;
#pragma unroll 8
  for (int j = 0; j < 64; ++j) {
    int h = hbase + j;
    float ang = (float)s * powf(10000.0f, -(float)(h & ~1) / (float)HH);
    float pe  = (h & 1) ? cosf(ang) : sinf(ang);
    float v   = acc[j] + pe;
    xo[j] = v;
    xb[j] = (__bf16)v;
  }
}

// ---------------- kernel 2: QKV projections (6 GEMMs) ----------------
// Weights pre-transposed: Wt [A x H].  Q,K stored [rows, A]; V stored
// transposed per batch [b][A][S] so the P@V B-operand is K-contiguous.
// Each wave computes a 16x64 output tile (4 accumulators, A reused 4x).
__global__ void k_qkv(const __bf16* __restrict__ xb,
                      const __bf16* __restrict__ w6t,
                      __bf16* __restrict__ qkv,
                      __bf16* __restrict__ vt) {
  int lane = threadIdx.x & 31;
  int wave = threadIdx.x >> 5;
  int m0 = blockIdx.x * 16;
  int y  = blockIdx.y;  // 0..5 = WQ1,WK1,WV1,WQ2,WK2,WV2
  const __bf16* Wt = w6t + (size_t)y * (AA * HH);
  bool isV = (y == 2 || y == 5);
  int nbase = wave * 64;

  v8f acc[4];
#pragma unroll
  for (int j = 0; j < 4; ++j) acc[j] = zero8();

  gemm_tile<HH / 32>(xb, HH, m0, Wt, HH, nbase, lane, acc);

  if (!isV) {
    __bf16* o = qkv + (size_t)y * (ROWS * AA);
#pragma unroll
    for (int j = 0; j < 4; ++j)
      store_c_bf16(o, AA, m0, nbase + j * 16, lane, acc[j]);
  } else {
    int b  = m0 >> 10;          // batch (tiles never cross batch bound)
    int s0 = m0 & (SS - 1);
    __bf16* ov = vt + (size_t)((y == 5 ? BB : 0) + b) * AA * SS;
    int mb = (lane & 16) ? 8 : 0;
#pragma unroll
    for (int j = 0; j < 4; ++j) {
      int n = nbase + j * 16 + (lane & 15);
#pragma unroll
      for (int r = 0; r < 8; ++r)
        ov[(size_t)n * SS + (s0 + mb + r)] = (__bf16)acc[j][r];
    }
  }
}

// ---------------- kernel 3: attention (per batch/branch/16-query tile) ----
// grid = (S/16, 2, B), block = 1 wave; LDS = 16x1024 f32 (64KB).
// Scores written f32; softmax packs unnormalized exp() as bf16 in-place
// (bf16 elem j overwrites f32 elem j/2 which is already consumed);
// 1/sum is stashed at f32 slot 512 of each row and folded into the P@V
// epilogue, so P@V is pure ds_load_b128 -> wmma with no conversions.
__global__ void __launch_bounds__(32) k_attn(const __bf16* __restrict__ qkv,
                                             const __bf16* __restrict__ vt,
                                             const float* __restrict__ mask,
                                             __bf16* __restrict__ zcat) {
  __shared__ __align__(16) float sc[16 * SS];  // 64 KB
  int lane = threadIdx.x;
  int m0 = blockIdx.x * 16;
  int br = blockIdx.y;
  int b  = blockIdx.z;

  const size_t mstride = ROWS * AA;
  const __bf16* Q  = qkv + (size_t)(br * 3 + 0) * mstride + (size_t)b * SS * AA;
  const __bf16* K  = qkv + (size_t)(br * 3 + 1) * mstride + (size_t)b * SS * AA;
  const __bf16* Vt = vt + (size_t)(br * BB + b) * AA * SS;  // [A x S]

  // scores = Q Kᵀ / 16 + mask_bias  (4 key-tiles per group, Q frag reused)
  for (int g = 0; g < SS / 64; ++g) {
    v8f acc[4];
#pragma unroll
    for (int j = 0; j < 4; ++j) acc[j] = zero8();

    gemm_tile<AA / 32>(Q, AA, m0, K, AA, g * 64, lane, acc);

    int mb = (lane & 16) ? 8 : 0;
#pragma unroll
    for (int j = 0; j < 4; ++j) {
      int n = g * 64 + j * 16 + (lane & 15);
      float bias = mask[(size_t)b * SS + n] * 1e6f - 1e6f;
#pragma unroll
      for (int r = 0; r < 8; ++r)
        sc[(mb + r) * SS + n] = acc[j][r] * 0.0625f + bias;
    }
  }
  __syncthreads();

  // masked softmax per row; pack unnormalized exp as bf16 in place
  if (lane < 16) {
    float* row = sc + lane * SS;
    bf16_alias* prow = (bf16_alias*)row;
    float mx = -1e30f;
    for (int j = 0; j < SS; ++j) mx = fmaxf(mx, row[j]);
    float sum = 0.0f;
    for (int j = 0; j < SS; ++j) {
      float e = expf(row[j] - mx);  // reads f32 j; write below hits f32 j/2
      sum += e;
      prow[j] = (__bf16)e;
    }
    row[512] = 1.0f / sum;          // bytes 2048.. : untouched by the pack
  }
  __syncthreads();

  // O = (Pexp @ V) * inv  : A = packed bf16 probs in LDS (row stride 2048)
  const __bf16* Pb = (const __bf16*)sc;
  __bf16* out = zcat + (size_t)((size_t)b * SS + m0) * (2 * AA) + (size_t)br * AA;
  for (int g = 0; g < AA / 64; ++g) {
    v8f acc[4];
#pragma unroll
    for (int j = 0; j < 4; ++j) acc[j] = zero8();

    gemm_tile<SS / 32>(Pb, 2 * SS, 0, Vt, SS, g * 64, lane, acc);

    int mb = (lane & 16) ? 8 : 0;
#pragma unroll
    for (int j = 0; j < 4; ++j) {
      int n = g * 64 + j * 16 + (lane & 15);
#pragma unroll
      for (int r = 0; r < 8; ++r) {
        float inv = sc[(size_t)(mb + r) * SS + 512];
        out[(size_t)(mb + r) * (2 * AA) + n] = (__bf16)(acc[j][r] * inv);
      }
    }
  }
}

// ---------------- kernel 4: W0 GEMM + residual + LayerNorm1 ----------------
__global__ void __launch_bounds__(32) k_w0_ln(const __bf16* __restrict__ zcat,
                                              const __bf16* __restrict__ w0t,
                                              const float* __restrict__ x,
                                              const float* __restrict__ g,
                                              const float* __restrict__ beta,
                                              float* __restrict__ h1,
                                              __bf16* __restrict__ h1b) {
  __shared__ __align__(16) float t[16 * HH];  // 32 KB
  int lane = threadIdx.x;
  int m0 = blockIdx.x * 16;

  for (int gn = 0; gn < HH / 64; ++gn) {
    v8f acc[4];
#pragma unroll
    for (int j = 0; j < 4; ++j) acc[j] = zero8();

    gemm_tile<HH / 32>(zcat, HH, m0, w0t, HH, gn * 64, lane, acc);

    int mb = (lane & 16) ? 8 : 0;
#pragma unroll
    for (int j = 0; j < 4; ++j) {
      int n = gn * 64 + j * 16 + (lane & 15);
#pragma unroll
      for (int r = 0; r < 8; ++r) t[(mb + r) * HH + n] = acc[j][r];
    }
  }
  __syncthreads();

  if (lane < 16) {
    int gr = m0 + lane;
    float* row = t + lane * HH;
    const float* xr = x + (size_t)gr * HH;
    float mean = 0.0f;
    for (int h = 0; h < HH; ++h) {
      float v = row[h] + xr[h];
      row[h] = v;
      mean += v;
    }
    mean *= (1.0f / HH);
    float var = 0.0f;
    for (int h = 0; h < HH; ++h) {
      float d = row[h] - mean;
      var += d * d;
    }
    var *= (1.0f / HH);
    float rs = rsqrtf(var + 1e-5f);
    float* ho = h1 + (size_t)gr * HH;
    __bf16* hb = h1b + (size_t)gr * HH;
    for (int h = 0; h < HH; ++h) {
      float y = (row[h] - mean) * rs * g[h] + beta[h];
      ho[h] = y;
      hb[h] = (__bf16)y;
    }
  }
}

// ---------------- kernel 5: FFN (relu(h1@Wf1+b1) @ Wf2 + b2) -> fz ----------
__global__ void __launch_bounds__(32) k_ffn(const __bf16* __restrict__ h1b,
                                            const __bf16* __restrict__ wf1t,
                                            const float* __restrict__ b1,
                                            const __bf16* __restrict__ wf2t,
                                            const float* __restrict__ b2,
                                            float* __restrict__ fz) {
  __shared__ __align__(16) __bf16 mid[16 * FF];  // 64 KB
  int lane = threadIdx.x;
  int m0 = blockIdx.x * 16;

  // mid = relu(h1 @ Wf1 + b1); Wf1 pre-transposed [F x H]
  for (int gn = 0; gn < FF / 64; ++gn) {
    __builtin_prefetch(wf1t + (size_t)(gn + 1) * 64 * HH, 0, 1);
    v8f acc[4];
#pragma unroll
    for (int j = 0; j < 4; ++j) acc[j] = zero8();

    gemm_tile<HH / 32>(h1b, HH, m0, wf1t, HH, gn * 64, lane, acc);

    int mb = (lane & 16) ? 8 : 0;
#pragma unroll
    for (int j = 0; j < 4; ++j) {
      int n = gn * 64 + j * 16 + (lane & 15);
      float bias = b1[n];
#pragma unroll
      for (int r = 0; r < 8; ++r)
        mid[(mb + r) * FF + n] = (__bf16)fmaxf(acc[j][r] + bias, 0.0f);
    }
  }
  __syncthreads();

  // fz = mid @ Wf2 + b2; Wf2 pre-transposed [H x F]
  for (int gn = 0; gn < HH / 64; ++gn) {
    __builtin_prefetch(wf2t + (size_t)(gn + 1) * 64 * FF, 0, 1);
    v8f acc[4];
#pragma unroll
    for (int j = 0; j < 4; ++j) acc[j] = zero8();

    gemm_tile<FF / 32>(mid, FF, 0, wf2t, FF, gn * 64, lane, acc);

    int mb = (lane & 16) ? 8 : 0;
#pragma unroll
    for (int j = 0; j < 4; ++j) {
      int n = gn * 64 + j * 16 + (lane & 15);
      float bias = b2[n];
#pragma unroll
      for (int r = 0; r < 8; ++r)
        fz[(size_t)(m0 + mb + r) * HH + n] = acc[j][r] + bias;
    }
  }
}

// ---------------- kernel 6: residual + LayerNorm2 + final projection --------
__global__ void k_ln2_fin(const float* __restrict__ fz,
                          const float* __restrict__ h1,
                          const float* __restrict__ g2,
                          const float* __restrict__ be2,
                          const float* __restrict__ wfin,
                          const float* __restrict__ bfin,
                          float* __restrict__ out) {
  int gr = blockIdx.x * blockDim.x + threadIdx.x;  // one token per thread
  const float* a = fz + (size_t)gr * HH;
  const float* b = h1 + (size_t)gr * HH;
  float mean = 0.0f;
  for (int h = 0; h < HH; ++h) mean += a[h] + b[h];
  mean *= (1.0f / HH);
  float var = 0.0f;
  for (int h = 0; h < HH; ++h) {
    float d = a[h] + b[h] - mean;
    var += d * d;
  }
  var *= (1.0f / HH);
  float rs = rsqrtf(var + 1e-5f);
  float dot = 0.0f;
  for (int h = 0; h < HH; ++h)
    dot += ((a[h] + b[h] - mean) * rs * g2[h] + be2[h]) * wfin[h];
  out[gr] = dot + bfin[0];
}

// ---------------- host launch ----------------
extern "C" void kernel_launch(void* const* d_in, const int* in_sizes, int n_in,
                              void* d_out, int out_size, void* d_ws, size_t ws_size,
                              hipStream_t stream) {
  (void)in_sizes; (void)n_in; (void)out_size; (void)ws_size;

  const int*   cate_x = (const int*)d_in[0];
  const float* cont_x = (const float*)d_in[1];
  const float* mask   = (const float*)d_in[2];
  /* d_in[3] targets unused by forward */
  const float* E_cate = (const float*)d_in[4];
  const float* W_proj = (const float*)d_in[5];
  const float* W0     = (const float*)d_in[12];
  const float* ln1_g  = (const float*)d_in[13];
  const float* ln1_b  = (const float*)d_in[14];
  const float* Wf1    = (const float*)d_in[15];
  const float* bf1    = (const float*)d_in[16];
  const float* Wf2    = (const float*)d_in[17];
  const float* bf2    = (const float*)d_in[18];
  const float* ln2_g  = (const float*)d_in[19];
  const float* ln2_b  = (const float*)d_in[20];
  const float* Wfin   = (const float*)d_in[21];
  const float* bfin   = (const float*)d_in[22];

  char* ws = (char*)d_ws;
  size_t off = 0;
  float*  x_f32  = (float*)(ws + off);  off += ROWS * HH * sizeof(float);
  __bf16* x_b16  = (__bf16*)(ws + off); off += ROWS * HH * sizeof(__bf16);
  __bf16* w6t    = (__bf16*)(ws + off); off += (size_t)6 * AA * HH * sizeof(__bf16);
  __bf16* w0t    = (__bf16*)(ws + off); off += (size_t)HH * HH * sizeof(__bf16);
  __bf16* wf1t   = (__bf16*)(ws + off); off += (size_t)FF * HH * sizeof(__bf16);
  __bf16* wf2t   = (__bf16*)(ws + off); off += (size_t)HH * FF * sizeof(__bf16);
  __bf16* qkv    = (__bf16*)(ws + off); off += (size_t)6 * ROWS * AA * sizeof(__bf16);
  __bf16* vt     = (__bf16*)(ws + off); off += (size_t)2 * BB * AA * SS * sizeof(__bf16);
  __bf16* zcat   = (__bf16*)(ws + off); off += ROWS * (2 * AA) * sizeof(__bf16);
  float*  h1     = (float*)(ws + off);  off += ROWS * HH * sizeof(float);
  __bf16* h1b    = (__bf16*)(ws + off); off += ROWS * HH * sizeof(__bf16);
  float*  fz     = (float*)(ws + off);  off += ROWS * HH * sizeof(float);

  // weight conversions to transposed bf16
  for (int i = 0; i < 6; ++i) {
    k_f32_to_bf16_T<<<(HH * AA + 255) / 256, 256, 0, stream>>>(
        (const float*)d_in[6 + i], w6t + (size_t)i * AA * HH, HH, AA);
  }
  k_f32_to_bf16_T<<<(HH * HH + 255) / 256, 256, 0, stream>>>(W0, w0t, HH, HH);
  k_f32_to_bf16_T<<<(HH * FF + 255) / 256, 256, 0, stream>>>(Wf1, wf1t, HH, FF);
  k_f32_to_bf16_T<<<(FF * HH + 255) / 256, 256, 0, stream>>>(Wf2, wf2t, FF, HH);

  k_embed_proj<<<ROWS / 32, 256, 0, stream>>>(cate_x, cont_x, E_cate, W_proj,
                                              x_f32, x_b16);
  k_qkv<<<dim3(ROWS / 16, 6), 128, 0, stream>>>(x_b16, w6t, qkv, vt);
  k_attn<<<dim3(SS / 16, 2, BB), 32, 0, stream>>>(qkv, vt, mask, zcat);
  k_w0_ln<<<ROWS / 16, 32, 0, stream>>>(zcat, w0t, x_f32, ln1_g, ln1_b, h1, h1b);
  k_ffn<<<ROWS / 16, 32, 0, stream>>>(h1b, wf1t, bf1, wf2t, bf2, fz);
  k_ln2_fin<<<ROWS / 256, 256, 0, stream>>>(fz, h1, ln2_g, ln2_b, Wfin, bfin,
                                            (float*)d_out);
}
